// SectorWarGNN_70368744178416
// MI455X (gfx1250) — matrix operand order repaired and checked by reading, not compiled, not run on hardware
//
#include <hip/hip_runtime.h>
#include <cstdint>
#include <cstddef>

typedef __attribute__((ext_vector_type(16))) _Float16 v16h;
typedef __attribute__((ext_vector_type(8)))  float    v8f;

#define DEV static __device__ __forceinline__

// ---------------- model dimensions ----------------
constexpr int T_   = 2048;
constexpr int N_   = 11;    // nodes
constexpr int E_   = 110;   // edges
constexpr int NF_  = 5;
constexpr int EF_  = 4;
constexpr int EED_ = 32;

// padded row strides (floats). stride mod 64 chosen so the 16 A-rows map to
// 16 distinct LDS banks (conflict-free A-fragment builds).
constexpr int SH_ = 68;    // 16 x 64 node buffer
constexpr int SA_ = 260;   // 16 x 256 buffers
constexpr int SF_ = 292;   // 16 x 288 fusion buffer
constexpr int GROWS_ = 12; // 11 rows + 1 dump row for clamped stores

// k-tiles of W_hh cached in LDS for the serial GRU phase (4 * 48 frags * 1KB)
constexpr int KTL_ = 4;

// =====================================================================
// WMMA helpers. B fragments pre-swizzled in workspace: per (kt,nt) frag,
// lane l owns 16 halves: k = kt*32 + (l>>4)*8 + (j<8 ? j : 16+j-8), n = l&15.
// A fragment (ISA 16-bit A layout): lane l: m=l&15, kb=(l>>4)*8;
// vgpr v<4 -> k=kb+2v+(j&1), v>=4 -> k=kb+16+2(v-4)+(j&1).
// =====================================================================
template <int KT>
DEV void load_afrags(const float* A, int lda, v16h (&a)[KT]) {
  const int lane = threadIdx.x & 31;
  const int m    = lane & 15;
  const int kb   = (lane >> 4) * 8;
#pragma unroll
  for (int kt = 0; kt < KT; ++kt) {
#pragma unroll
    for (int j = 0; j < 16; ++j) {
      int v = j >> 1;
      int k = kt * 32 + kb + ((v < 4) ? (2 * v) : (16 + 2 * (v - 4))) + (j & 1);
      a[kt][j] = (_Float16)A[m * lda + k];
    }
  }
}

template <int KT>
DEV v8f mm_frags(const v16h (&a)[KT], const v16h* B, int ntiles, int nt, v8f c) {
  const int lane = threadIdx.x & 31;
#pragma unroll
  for (int kt = 0; kt < KT; ++kt) {
    v16h b = B[((size_t)(kt * ntiles + nt)) * 32 + lane];
    c = __builtin_amdgcn_wmma_f32_16x16x32_f16(false, a[kt], false, b, (short)0,
                                               c, false, false);
  }
  return c;
}

// bias folded into accumulator init (WMMA accumulates onto C for free).
// bias must be non-null (avoids a per-tile branch ladder).
DEV v8f init_c(const float* __restrict__ bias, int nt) {
  const int n = (threadIdx.x & 31) & 15;
  float bv = bias[nt * 16 + n];
  v8f c;
#pragma unroll
  for (int r = 0; r < 8; ++r) c[r] = bv;
  return c;
}

// C/D layout: vgpr r: lanes 0-15 -> M=r, lanes 16-31 -> M=r+8; N = lane&15.
// Branchless: rows beyond maxrow are clamped to the dump row `maxrow`.
DEV void store_tile(float* O, int ldo, int nt, int maxrow, v8f c, bool relu) {
  const int lane = threadIdx.x & 31;
  const int n    = lane & 15;
  const int mh   = (lane >> 4) * 8;
#pragma unroll
  for (int r = 0; r < 8; ++r) {
    int m = mh + r;
    m = (m > maxrow) ? maxrow : m;
    float v = c[r];
    if (relu) v = fmaxf(v, 0.0f);
    O[m * ldo + nt * 16 + n] = v;
  }
}

// =====================================================================
// Weight prep: fp32 row-major (K x N) -> f16 WMMA B fragments
// =====================================================================
__global__ void prep_weights(const float* __restrict__ src,
                             _Float16* __restrict__ dst, int K, int N) {
  int idx = blockIdx.x * blockDim.x + threadIdx.x;
  if (idx >= K * N) return;
  int k = idx / N, n = idx % N;
  int kt = k >> 5, nt = n >> 4;
  int lane = (n & 15) | (((k >> 3) & 1) << 4);
  int j = (k & 7) | (((k >> 4) & 1) << 3);
  dst[(((size_t)(kt * (N >> 4) + nt)) * 32 + lane) * 16 + j] =
      (_Float16)src[(size_t)k * N + n];
}

// =====================================================================
// GAT aggregate: edge scores, global softmax over all E edges,
// alpha-weighted scatter-add via LDS float atomics (ds_add_f32), relu.
// hW: (16 x SA_) LDS, out: (16 x SA_) LDS (rows 11..15 end up zero).
// =====================================================================
DEV void gat_aggregate(const float* hW, float* out, const float* s_ea,
                       const int* s_src, const int* s_dst, const float* Wa,
                       float bab, float* s_asrc, float* s_adst, float* s_logit,
                       float* s_red) {
  const int tid = threadIdx.x;
  if (tid < N_) {
    float as = 0.f, ad = 0.f;
    for (int j = 0; j < 256; ++j) {
      float v = hW[tid * SA_ + j];
      as += v * Wa[j];
      ad += v * Wa[256 + j];
    }
    s_asrc[tid] = as;
    s_adst[tid] = ad;
  }
  __syncthreads();
  for (int e = tid; e < E_; e += blockDim.x) {
    float lg = s_asrc[s_src[e]] + s_adst[s_dst[e]] + bab;
    for (int j = 0; j < 16; ++j) lg += s_ea[e * 16 + j] * Wa[512 + j];
    s_logit[e] = lg;
  }
  __syncthreads();
  if (tid == 0) {
    float m = -3.0e38f;
    for (int e = 0; e < E_; ++e) m = fmaxf(m, s_logit[e]);
    s_red[0] = m;
  }
  __syncthreads();
  float mx = s_red[0];
  for (int e = tid; e < E_; e += blockDim.x) s_logit[e] = __expf(s_logit[e] - mx);
  __syncthreads();
  if (tid == 0) {
    float s = 0.f;
    for (int e = 0; e < E_; ++e) s += s_logit[e];
    s_red[1] = s;
  }
  __syncthreads();
  float inv = 1.0f / s_red[1];
  for (int i = tid; i < 16 * SA_; i += blockDim.x) out[i] = 0.f;
  __syncthreads();
  for (int idx = tid; idx < E_ * 256; idx += blockDim.x) {
    int e = idx >> 8, j = idx & 255;
    atomicAdd(&out[s_dst[e] * SA_ + j],
              s_logit[e] * inv * hW[s_src[e] * SA_ + j]);
  }
  __syncthreads();
  for (int i = tid; i < 16 * SA_; i += blockDim.x) out[i] = fmaxf(out[i], 0.f);
  __syncthreads();
}

// =====================================================================
// Phase 1 (parallel over T): node/edge proj, GAT1, GAT2, LayerNorm,
// gi = x2_ln @ W_ih + b_ih  -> global workspace (12 padded rows)
// =====================================================================
__global__ void __launch_bounds__(128)
phase1(const float* __restrict__ x, const float* __restrict__ eattr,
       const int* __restrict__ eidx, const float* __restrict__ W_np,
       const float* __restrict__ b_np, const float* __restrict__ bn_g,
       const float* __restrict__ bn_b, const float* __restrict__ We1,
       const float* __restrict__ be1, const float* __restrict__ We2,
       const float* __restrict__ be2, const float* __restrict__ bg1,
       const float* __restrict__ Wa1, const float* __restrict__ ba1,
       const float* __restrict__ bg2, const float* __restrict__ Wa2,
       const float* __restrict__ ba2, const float* __restrict__ ln_g,
       const float* __restrict__ ln_b, const float* __restrict__ b_ih,
       const v16h* __restrict__ fWg1, const v16h* __restrict__ fWg2,
       const v16h* __restrict__ fWih, float* __restrict__ gi_out) {
  __shared__ float s_h[16 * SH_];
  __shared__ float s_A[16 * SA_];
  __shared__ float s_B[16 * SA_];
  __shared__ float s_ea[E_ * 16];
  __shared__ float s_asrc[16], s_adst[16];
  __shared__ float s_logit[128];
  __shared__ float s_red[2];
  __shared__ int   s_src[E_], s_dst[E_];

  const int t = blockIdx.x;
  const int tid = threadIdx.x;
  const int wave = tid >> 5, nwaves = blockDim.x >> 5;

  for (int e = tid; e < E_; e += blockDim.x) {
    s_src[e] = eidx[e];
    s_dst[e] = eidx[E_ + e];
  }

  // ---- node_proj: relu(xt @ W_np + b_np) -> eval BatchNorm ----
  const float* xt = x + (size_t)t * N_ * NF_;
  const float bn_inv = rsqrtf(1.0f + 1e-5f);
  for (int i = tid; i < 16 * SH_; i += blockDim.x) {
    int m = i / SH_, n = i % SH_;
    float v = 0.f;
    if (m < N_ && n < 64) {
      v = b_np[n];
      for (int k = 0; k < NF_; ++k) v += xt[m * NF_ + k] * W_np[k * 64 + n];
      v = fmaxf(v, 0.f);
      v = bn_g[n] * (v * bn_inv) + bn_b[n];
    }
    s_h[i] = v;
  }

  // ---- edge_proj: relu(relu(eat@We1+be1)@We2+be2) ----
  const float* et = eattr + (size_t)t * E_ * EF_;
  for (int e = tid; e < E_; e += blockDim.x) {
    float t1[32];
    for (int n = 0; n < 32; ++n) {
      float v = be1[n];
      for (int k = 0; k < EF_; ++k) v += et[e * EF_ + k] * We1[k * 32 + n];
      t1[n] = fmaxf(v, 0.f);
    }
    for (int n = 0; n < 16; ++n) {
      float v = be2[n];
      for (int k = 0; k < 32; ++k) v += t1[k] * We2[k * 16 + n];
      s_ea[e * 16 + n] = fmaxf(v, 0.f);
    }
  }
  __syncthreads();

  // ---- GAT1: hW = s_h(16x64) @ Wg1(64x256) + bg1 ----
  {
    v16h af[2];
    load_afrags<2>(s_h, SH_, af);
    for (int nt = wave; nt < 16; nt += nwaves) {
      v8f c = init_c(bg1, nt);
      c = mm_frags<2>(af, fWg1, 16, nt, c);
      store_tile(s_A, SA_, nt, 15, c, false);
    }
  }
  __syncthreads();
  gat_aggregate(s_A, s_B, s_ea, s_src, s_dst, Wa1, ba1[0], s_asrc, s_adst,
                s_logit, s_red);  // s_B = x1 (relu, rows 11..15 zero)

  // ---- GAT2: hW2 = x1(16x256) @ Wg2(256x256) + bg2 ----
  {
    v16h af[8];
    load_afrags<8>(s_B, SA_, af);
    for (int nt = wave; nt < 16; nt += nwaves) {
      v8f c = init_c(bg2, nt);
      c = mm_frags<8>(af, fWg2, 16, nt, c);
      store_tile(s_A, SA_, nt, 15, c, false);
    }
  }
  __syncthreads();
  gat_aggregate(s_A, s_B, s_ea, s_src, s_dst, Wa2, ba2[0], s_asrc, s_adst,
                s_logit, s_red);  // s_B = x2 (relu)

  // ---- LayerNorm rows -> s_A (rows 11..15 zeroed) ----
  if (tid < 16) {
    if (tid < N_) {
      float mu = 0.f;
      for (int j = 0; j < 256; ++j) mu += s_B[tid * SA_ + j];
      mu *= (1.0f / 256.0f);
      float var = 0.f;
      for (int j = 0; j < 256; ++j) {
        float d = s_B[tid * SA_ + j] - mu;
        var += d * d;
      }
      var *= (1.0f / 256.0f);
      float inv = rsqrtf(var + 1e-5f);
      for (int j = 0; j < 256; ++j)
        s_A[tid * SA_ + j] = ln_g[j] * (s_B[tid * SA_ + j] - mu) * inv + ln_b[j];
    } else {
      for (int j = 0; j < 256; ++j) s_A[tid * SA_ + j] = 0.f;
    }
  }
  __syncthreads();

  // ---- gi = x2_ln(16x256) @ W_ih(256x768) + b_ih -> global (12 rows) ----
  float* git = gi_out + (size_t)t * GROWS_ * 768;
  {
    v16h af[8];
    load_afrags<8>(s_A, SA_, af);
    for (int nt = wave; nt < 48; nt += nwaves) {
      v8f c = init_c(b_ih, nt);
      c = mm_frags<8>(af, fWih, 48, nt, c);
      store_tile(git, 768, nt, N_, c, false);  // rows >=11 -> dump row 11
    }
  }
}

// =====================================================================
// Phase 2 (single persistent workgroup): sequential GRU over T.
// Per step: gh = h(16x256) @ W_hh(256x768) + b_hh (8 waves, 6 n-tiles each,
// A-frags built once per step). First KTL_ k-tiles of W_hh are cached in
// LDS (192KB of the WGP's 320KB) so most B-frag traffic is ds_load instead
// of L2 round-trips on the serial critical path.
// =====================================================================
__global__ void __launch_bounds__(256)
phase2(const float* __restrict__ hx, const v16h* __restrict__ fWhh,
       const float* __restrict__ b_hh, const float* __restrict__ gi,
       float* __restrict__ hseq) {
  __shared__ float s_h[16 * SA_];
  __shared__ float s_gh[GROWS_ * 768];
  __shared__ __align__(32) uint4 s_whh4[KTL_ * 48 * 32 * 2];  // 192KB
  const int tid = threadIdx.x;
  const int wave = tid >> 5, nwaves = blockDim.x >> 5;
  const int lane = tid & 31;

  // one-time copy: k-tiles 0..KTL_-1 of W_hh fragments into LDS
  {
    const uint4* srcq = (const uint4*)fWhh;
    for (int i = tid; i < KTL_ * 48 * 32 * 2; i += blockDim.x)
      s_whh4[i] = srcq[i];
  }
  for (int i = tid; i < 16 * SA_; i += blockDim.x) s_h[i] = 0.f;
  __syncthreads();
  for (int i = tid; i < N_ * 256; i += blockDim.x)
    s_h[(i >> 8) * SA_ + (i & 255)] = hx[i];
  __syncthreads();

  const v16h* sWhh = (const v16h*)s_whh4;

  for (int t = 0; t < T_; ++t) {
    const float* git = gi + (size_t)t * GROWS_ * 768;
    if (t + 1 < T_) {  // warm caches for next step's gi (global_prefetch)
      const float* nx = gi + (size_t)(t + 1) * GROWS_ * 768;
      for (int i = tid * 16; i < N_ * 768; i += blockDim.x * 16)
        __builtin_prefetch(&nx[i], 0, 1);
    }
    {
      v16h af[8];
      load_afrags<8>(s_h, SA_, af);
      for (int nt = wave; nt < 48; nt += nwaves) {
        v8f c = init_c(b_hh, nt);
#pragma unroll
        for (int kt = 0; kt < KTL_; ++kt) {  // LDS-resident k-tiles
          v16h b = sWhh[(kt * 48 + nt) * 32 + lane];
          c = __builtin_amdgcn_wmma_f32_16x16x32_f16(false, af[kt], false, b,
                                                     (short)0, c, false, false);
        }
#pragma unroll
        for (int kt = KTL_; kt < 8; ++kt) {  // L2-resident k-tiles
          v16h b = fWhh[((size_t)(kt * 48 + nt)) * 32 + lane];
          c = __builtin_amdgcn_wmma_f32_16x16x32_f16(false, af[kt], false, b,
                                                     (short)0, c, false, false);
        }
        store_tile(s_gh, 768, nt, N_, c, false);  // dump row 11
      }
    }
    __syncthreads();
    float* ht = hseq + (size_t)t * GROWS_ * 256;
    for (int idx = tid; idx < N_ * 256; idx += blockDim.x) {
      int m = idx >> 8, j = idx & 255;
      float ir = git[m * 768 + j];
      float iz = git[m * 768 + 256 + j];
      float in = git[m * 768 + 512 + j];
      float hr = s_gh[m * 768 + j];
      float hz = s_gh[m * 768 + 256 + j];
      float hn = s_gh[m * 768 + 512 + j];
      float r = 1.0f / (1.0f + __expf(-(ir + hr)));
      float z = 1.0f / (1.0f + __expf(-(iz + hz)));
      float n = tanhf(in + r * hn);
      float hnew = (1.0f - z) * n + z * s_h[m * SA_ + j];
      s_h[m * SA_ + j] = hnew;
      ht[idx] = hnew;
    }
    __syncthreads();
  }
}

// =====================================================================
// Phase 3 (parallel over T): event fusion + predictor MLP.
// =====================================================================
__global__ void __launch_bounds__(128)
phase3(const float* __restrict__ hseq, const float* __restrict__ evemb,
       const v16h* __restrict__ fWf, const float* __restrict__ bf,
       const v16h* __restrict__ fWp1, const float* __restrict__ bp1,
       const float* __restrict__ Wp2, const float* __restrict__ bp2,
       float* __restrict__ out) {
  __shared__ float s_A[16 * SF_];
  __shared__ float s_B[16 * SA_];
  const int t = blockIdx.x;
  const int tid = threadIdx.x;
  const int wave = tid >> 5, nwaves = blockDim.x >> 5;

  const float* h = hseq + (size_t)t * GROWS_ * 256;
  const float* evt = evemb + (size_t)t * EED_;
  for (int i = tid; i < 16 * SF_; i += blockDim.x) {
    int m = i / SF_, j = i % SF_;
    float v = 0.f;
    if (m < N_ && j < 288) v = (j < 256) ? h[m * 256 + j] : evt[j - 256];
    s_A[i] = v;
  }
  __syncthreads();

  // fused = relu([h, ev](16x288) @ Wf(288x256) + bf)
  {
    v16h af[9];
    load_afrags<9>(s_A, SF_, af);
    for (int nt = wave; nt < 16; nt += nwaves) {
      v8f c = init_c(bf, nt);
      c = mm_frags<9>(af, fWf, 16, nt, c);
      store_tile(s_B, SA_, nt, 15, c, true);
    }
  }
  __syncthreads();

  // p1 = relu(fused(16x256) @ Wp1(256x128) + bp1) -> reuse s_A (ld 128)
  {
    v16h af[8];
    load_afrags<8>(s_B, SA_, af);
    for (int nt = wave; nt < 8; nt += nwaves) {
      v8f c = init_c(bp1, nt);
      c = mm_frags<8>(af, fWp1, 8, nt, c);
      store_tile(s_A, 128, nt, 15, c, true);
    }
  }
  __syncthreads();

  if (tid < N_) {
    float acc = bp2[0];
    for (int j = 0; j < 128; ++j) acc += s_A[tid * 128 + j] * Wp2[j];
    out[(size_t)t * N_ + tid] = acc;
  }
}

// =====================================================================
// Host launcher
// =====================================================================
extern "C" void kernel_launch(void* const* d_in, const int* in_sizes, int n_in,
                              void* d_out, int out_size, void* d_ws,
                              size_t ws_size, hipStream_t stream) {
  const float* x     = (const float*)d_in[0];
  const float* eattr = (const float*)d_in[1];
  const float* evemb = (const float*)d_in[2];
  const float* hx    = (const float*)d_in[3];
  const int*   eidx  = (const int*)d_in[4];
  const float* W_np = (const float*)d_in[5];  const float* b_np = (const float*)d_in[6];
  const float* bn_g = (const float*)d_in[7];  const float* bn_b = (const float*)d_in[8];
  const float* We1  = (const float*)d_in[9];  const float* be1  = (const float*)d_in[10];
  const float* We2  = (const float*)d_in[11]; const float* be2  = (const float*)d_in[12];
  const float* Wg1  = (const float*)d_in[13]; const float* bg1  = (const float*)d_in[14];
  const float* Wa1  = (const float*)d_in[15]; const float* ba1  = (const float*)d_in[16];
  const float* Wg2  = (const float*)d_in[17]; const float* bg2  = (const float*)d_in[18];
  const float* Wa2  = (const float*)d_in[19]; const float* ba2  = (const float*)d_in[20];
  const float* ln_g = (const float*)d_in[21]; const float* ln_b = (const float*)d_in[22];
  const float* W_ih = (const float*)d_in[23]; const float* W_hh = (const float*)d_in[24];
  const float* b_ih = (const float*)d_in[25]; const float* b_hh = (const float*)d_in[26];
  const float* Wf   = (const float*)d_in[27]; const float* bf   = (const float*)d_in[28];
  const float* Wp1  = (const float*)d_in[29]; const float* bp1  = (const float*)d_in[30];
  const float* Wp2  = (const float*)d_in[31]; const float* bp2  = (const float*)d_in[32];

  char* ws = (char*)d_ws;
  size_t off = 0;
  auto take = [&](size_t bytes) {
    size_t r = off;
    off = (off + bytes + 255) & ~(size_t)255;
    return r;
  };
  _Float16* fWg1 = (_Float16*)(ws + take((size_t)64 * 256 * 2));
  _Float16* fWg2 = (_Float16*)(ws + take((size_t)256 * 256 * 2));
  _Float16* fWih = (_Float16*)(ws + take((size_t)256 * 768 * 2));
  _Float16* fWhh = (_Float16*)(ws + take((size_t)256 * 768 * 2));
  _Float16* fWf  = (_Float16*)(ws + take((size_t)288 * 256 * 2));
  _Float16* fWp1 = (_Float16*)(ws + take((size_t)256 * 128 * 2));
  float* gi   = (float*)(ws + take((size_t)T_ * GROWS_ * 768 * 4));
  float* hseq = (float*)(ws + take((size_t)T_ * GROWS_ * 256 * 4));

  auto prep = [&](const float* src, _Float16* dst, int K, int N) {
    int total = K * N;
    prep_weights<<<(total + 255) / 256, 256, 0, stream>>>(src, dst, K, N);
  };
  prep(Wg1, fWg1, 64, 256);
  prep(Wg2, fWg2, 256, 256);
  prep(W_ih, fWih, 256, 768);
  prep(W_hh, fWhh, 256, 768);
  prep(Wf, fWf, 288, 256);
  prep(Wp1, fWp1, 256, 128);

  phase1<<<T_, 128, 0, stream>>>(x, eattr, eidx, W_np, b_np, bn_g, bn_b, We1,
                                 be1, We2, be2, bg1, Wa1, ba1, bg2, Wa2, ba2,
                                 ln_g, ln_b, b_ih, (const v16h*)fWg1,
                                 (const v16h*)fWg2, (const v16h*)fWih, gi);

  phase2<<<1, 256, 0, stream>>>(hx, (const v16h*)fWhh, b_hh, gi, hseq);

  phase3<<<T_, 128, 0, stream>>>(hseq, evemb, (const v16h*)fWf, bf,
                                 (const v16h*)fWp1, bp1, Wp2, bp2,
                                 (float*)d_out);
}
